// ScaledDotProductAttention_33706903339367
// MI455X (gfx1250) — matrix operand order repaired and checked by reading, not compiled, not run on hardware
//
#include <hip/hip_runtime.h>
#include <hip/hip_bf16.h>
#include <math.h>

typedef __attribute__((ext_vector_type(16))) __bf16 v16bf;
typedef __attribute__((ext_vector_type(8)))  __bf16 v8bf;
typedef __attribute__((ext_vector_type(8)))  float  v8f;
typedef __attribute__((ext_vector_type(4)))  unsigned int u32x4;
typedef __attribute__((ext_vector_type(8)))  int i32x8;
typedef __attribute__((ext_vector_type(4)))  int i32x4;

#if defined(__has_builtin)
#if __has_builtin(__builtin_amdgcn_tensor_load_to_lds) && \
    __has_builtin(__builtin_amdgcn_s_wait_tensorcnt)
#define HAVE_TDM 1
#endif
#endif
#ifndef HAVE_TDM
#define HAVE_TDM 0
#endif

union V16U { v16bf v; v8bf h[2]; };

__device__ __forceinline__ v16bf make_v16(v8bf lo, v8bf hi) {
  V16U u; u.h[0] = lo; u.h[1] = hi; return u.v;
}

__device__ __forceinline__ v8f v8f_zero() {
  v8f z;
#pragma unroll
  for (int i = 0; i < 8; ++i) z[i] = 0.0f;
  return z;
}

__device__ __forceinline__ unsigned short f32_to_bf16_rne(float f) {
  union { float f; unsigned int u; } x; x.f = f;
  unsigned int u = x.u;
  u += 0x7FFFu + ((u >> 16) & 1u);
  return (unsigned short)(u >> 16);
}

__device__ __forceinline__ v8f wmma_bf16(v16bf a, v16bf b, v8f c) {
  return __builtin_amdgcn_wmma_f32_16x16x32_bf16(false, a, false, b,
                                                 (short)0, c, false, false);
}

#if HAVE_TDM
// ---------------------------------------------------------------------------
// TDM: DMA a 2D bf16 tile (tile_w elems x tile_h rows, row stride in elems)
// from global memory into LDS, inserting pad_amount DWORD padding every
// (1<<pad_interval) 8-byte units so the LDS row stride matches our layout.
// D# packing per CDNA5 ISA (2D: groups 2/3 = 0).
// Wave-level op (EXEC ignored); tracked with TENSORcnt.
// This toolchain exposes the 6-arg builtin: (g0, g1, g2, g3, g_ext, cpol).
// ---------------------------------------------------------------------------
__device__ __forceinline__ void tdm_load_2d_bf16(
    unsigned int lds_off, const void* gptr, unsigned int tile_w,
    unsigned int tile_h, unsigned long long row_stride_elems,
    unsigned int pad_interval_code, unsigned int pad_amount_code) {
  unsigned long long ga = (unsigned long long)gptr;
  u32x4 g0;
  g0[0] = 1u;                                    // count=1 user descriptor
  g0[1] = lds_off;                               // lds_addr (bytes)
  g0[2] = (unsigned int)ga;                      // global_addr[31:0]
  g0[3] = (unsigned int)((ga >> 32) & 0x01FFFFFFull) | (2u << 30); // [56:32]|type=2
  i32x8 g1;
  g1[0] = (int)((1u << 16) |                     // data_size = 2 bytes
                (1u << 20) |                     // pad_enable
                (pad_interval_code << 22) |      // pad every 2^c 8B units
                (pad_amount_code << 25));        // pad (c+1) DWORDs
  g1[1] = (int)((tile_w & 0xFFFFu) << 16);       // tensor_dim0[15:0] @ bits63:48
  g1[2] = (int)((tile_w >> 16) | ((tile_h & 0xFFFFu) << 16)); // dim0 hi | dim1 lo
  g1[3] = (int)((tile_h >> 16) | (tile_w << 16));             // dim1 hi | tile_dim0
  g1[4] = (int)(tile_h & 0xFFFFu);               // tile_dim1, tile_dim2=0
  g1[5] = (int)(unsigned int)row_stride_elems;   // tensor_dim0_stride lo32
  g1[6] = (int)(unsigned int)((row_stride_elems >> 32) & 0xFFFFull); // stride hi16
  g1[7] = 0;                                     // tensor_dim1_stride = 0 (2D)
  i32x4 z4 = {0, 0, 0, 0};
  i32x8 z8 = {0, 0, 0, 0, 0, 0, 0, 0};
  __builtin_amdgcn_tensor_load_to_lds(g0, g1, z4, z4, z8, 0);
}

__device__ __forceinline__ unsigned int lds_offset_of(const void* p) {
  return (unsigned int)(unsigned long long)p;     // generic LDS ptr: low 32 = offset
}
#endif

// ---------------------------------------------------------------------------
// Tiled GEMM: C[M,N] = A[M,K] * B[K,N], B is f32 weights (converted to bf16).
// ABF16: A is bf16 (workspace ctx, staged via TDM) else f32 (converted).
// HEADSTORE: write bf16 to per-head layout [b, h, pos, 64]; else plain f32.
// Block: 256 threads (8 waves); tile 128x128x32; wave sub-tile 32x64.
// ---------------------------------------------------------------------------
template <bool ABF16, bool HEADSTORE>
__global__ __launch_bounds__(256)
void gemm_bf16_wmma(const void* __restrict__ Ain, const float* __restrict__ Bw,
                    void* __restrict__ Cout, int M, int Kdim, int N, int len) {
  (void)M; (void)len;
  constexpr int BM = 128, BN = 128, BK = 32, LSTR = BK + 8; // 40 us = 80B (16B mult)
  constexpr int Hc = 16;
  __shared__ __align__(16) unsigned short As[BM][LSTR];
  __shared__ __align__(16) unsigned short Bt[BN][LSTR]; // transposed [n][k]

  const int tid  = threadIdx.x;
  const int lane = tid & 31;
  const int wid  = tid >> 5;
  const int wm   = wid & 3;   // 4 wave-rows of 32
  const int wn   = wid >> 2;  // 2 wave-cols of 64
  const int m0   = blockIdx.y * BM;
  const int n0   = blockIdx.x * BN;

  v8f acc[2][4];
#pragma unroll
  for (int i = 0; i < 2; ++i)
#pragma unroll
    for (int j = 0; j < 4; ++j) acc[i][j] = v8f_zero();

  const int asel = (lane & 16) ? 8 : 0;   // A-fragment K offset
  const int bsel = (lane & 16) ? 8 : 0;   // B-fragment K offset

  for (int k0 = 0; k0 < Kdim; k0 += BK) {
    __syncthreads();
    // ---- stage A tile (128x32) into LDS as bf16 ----
    if constexpr (ABF16) {
      const unsigned short* A = (const unsigned short*)Ain;
#if HAVE_TDM
      if (wid == 0) {
        // rows of 32 bf16 = 64B = 8x8B units -> pad_interval=3; pad 4 DW -> 3
        tdm_load_2d_bf16(lds_offset_of(&As[0][0]),
                         A + (size_t)m0 * Kdim + k0, BK, BM,
                         (unsigned long long)Kdim, 3u, 3u);
      }
#else
#pragma unroll
      for (int it = 0; it < 2; ++it) {
        int slot = tid + it * 256;       // 512 slots of 8 elems
        int row = slot >> 2, c8 = slot & 3;
        uint4 d = *(const uint4*)(A + (size_t)(m0 + row) * Kdim + k0 + c8 * 8);
        *(uint4*)&As[row][c8 * 8] = d;
      }
#endif
    } else {
      const float* A = (const float*)Ain;
#pragma unroll
      for (int it = 0; it < 4; ++it) {
        int slot = tid + it * 256;       // 1024 slots of 4 elems
        int row = slot >> 3, c4 = slot & 7;
        float4 d = *(const float4*)(A + (size_t)(m0 + row) * Kdim + k0 + c4 * 4);
        unsigned short* p = &As[row][c4 * 4];
        p[0] = f32_to_bf16_rne(d.x); p[1] = f32_to_bf16_rne(d.y);
        p[2] = f32_to_bf16_rne(d.z); p[3] = f32_to_bf16_rne(d.w);
      }
    }
    // ---- stage B tile (32x128) transposed into LDS as bf16 ----
#pragma unroll
    for (int it = 0; it < 4; ++it) {
      int slot = tid + it * 256;         // 1024 slots of 4 elems
      int kr = slot >> 5, c4 = slot & 31;
      float4 d = *(const float4*)(Bw + (size_t)(k0 + kr) * N + n0 + c4 * 4);
      Bt[c4 * 4 + 0][kr] = f32_to_bf16_rne(d.x);
      Bt[c4 * 4 + 1][kr] = f32_to_bf16_rne(d.y);
      Bt[c4 * 4 + 2][kr] = f32_to_bf16_rne(d.z);
      Bt[c4 * 4 + 3][kr] = f32_to_bf16_rne(d.w);
    }
#if HAVE_TDM
    if constexpr (ABF16) {
      if (wid == 0) __builtin_amdgcn_s_wait_tensorcnt(0);
    }
#endif
    __syncthreads();

    // ---- WMMA: 2 (M) x 4 (N) 16x16 tiles per wave, K=32 per iteration ----
    v16bf afrag[2];
#pragma unroll
    for (int i = 0; i < 2; ++i) {
      const unsigned short* ap = &As[wm * 32 + i * 16 + (lane & 15)][asel];
      afrag[i] = make_v16(*(const v8bf*)ap, *(const v8bf*)(ap + 16));
    }
#pragma unroll
    for (int j = 0; j < 4; ++j) {
      const unsigned short* bp = &Bt[wn * 64 + j * 16 + (lane & 15)][bsel];
      v16bf bfrag = make_v16(*(const v8bf*)bp, *(const v8bf*)(bp + 16));
#pragma unroll
      for (int i = 0; i < 2; ++i)
        acc[i][j] = wmma_bf16(afrag[i], bfrag, acc[i][j]);
    }
  }

  // ---- store C ----
  const int rbase = (lane & 16) ? 8 : 0;
  const int cl = lane & 15;
#pragma unroll
  for (int i = 0; i < 2; ++i) {
#pragma unroll
    for (int j = 0; j < 4; ++j) {
#pragma unroll
      for (int r = 0; r < 8; ++r) {
        int gm = m0 + wm * 32 + i * 16 + rbase + r;
        int gn = n0 + wn * 64 + j * 16 + cl;
        float v = acc[i][j][r];
        if constexpr (HEADSTORE) {
          unsigned short* Cb = (unsigned short*)Cout;
          int bb = gm / len, pos = gm % len;
          int hh = gn >> 6, hd = gn & 63;
          Cb[(((size_t)bb * Hc + hh) * len + pos) * 64 + hd] = f32_to_bf16_rne(v);
        } else {
          float* Cf = (float*)Cout;
          Cf[(size_t)gm * N + gn] = v;
        }
      }
    }
  }
}

// ---------------------------------------------------------------------------
// Flash attention with post-softmax multiplicative indicator mask.
// Block: 256 threads = 8 waves; each wave owns a 16-query tile (128 q/block).
// Grid: B * H * (NQ/128). Q/K/V are bf16 in [b,h,pos,64] layout.
// K tile and per-wave Q tiles staged by the Tensor Data Mover (padded rows);
// V staged cooperatively transposed (TDM cannot transpose).
// Softmax denominator is UNMASKED (matches reference); indicator scales probs
// only before the P*V matmul.
// ---------------------------------------------------------------------------
__global__ __launch_bounds__(256)
void flash_attn_bf16(const unsigned short* __restrict__ Qw,
                     const unsigned short* __restrict__ Kw,
                     const unsigned short* __restrict__ Vw,
                     const float* __restrict__ indic,
                     unsigned short* __restrict__ ctxw,
                     int B_, int H_, int NQ_, int S_) {
  constexpr int HD = 64, LSTR = 72; // 72 us = 144B (16B multiple)
  __shared__ __align__(16) unsigned short Ks[64][LSTR];      // [key][hd]
  __shared__ __align__(16) unsigned short Vt[64][LSTR];      // [hd][key]
  __shared__ __align__(16) unsigned short Qs[8][16][LSTR];   // per-wave Q tile
  __shared__ __align__(16) unsigned short Ps[8][16][LSTR];   // per-wave probs
  __shared__ float inds[64];

  const int tid = threadIdx.x, lane = tid & 31, wid = tid >> 5;
  const int nqc = NQ_ / 128;
  const int qc = blockIdx.x % nqc;
  const int bh = blockIdx.x / nqc;
  const int h = bh % H_, b = bh / H_;
  const int q0 = qc * 128 + wid * 16;
  const size_t qbase = ((size_t)(b * H_ + h) * NQ_ + q0) * HD;
  const size_t kvbase = (size_t)(b * H_ + h) * S_ * HD;

  // stage this wave's 16x64 Q tile (bf16) into LDS
#if HAVE_TDM
  // rows of 64 bf16 = 128B = 16x8B units -> pad_interval=4; pad 4 DW -> 3
  tdm_load_2d_bf16(lds_offset_of(&Qs[wid][0][0]), Qw + qbase, HD, 16,
                   (unsigned long long)HD, 4u, 3u);
  __builtin_amdgcn_s_wait_tensorcnt(0);
#else
#pragma unroll
  for (int it = 0; it < 4; ++it) {
    int slot = lane + it * 32;          // 128 slots of 8
    int row = slot >> 3, c8 = slot & 7;
    uint4 d = *(const uint4*)(Qw + qbase + (size_t)row * HD + c8 * 8);
    *(uint4*)&Qs[wid][row][c8 * 8] = d;
  }
#endif

  const int asel = (lane & 16) ? 8 : 0;
  const int bsel = (lane & 16) ? 8 : 0;

  // hoist Q A-fragments for entire S loop (same-wave DS ordering: no barrier)
  v16bf aq[2];
#pragma unroll
  for (int kc = 0; kc < 2; ++kc) {
    const unsigned short* p = &Qs[wid][lane & 15][kc * 32 + asel];
    aq[kc] = make_v16(*(const v8bf*)p, *(const v8bf*)(p + 16));
  }

  float mrow[8], lrow[8];
  v8f acc[4];
#pragma unroll
  for (int r = 0; r < 8; ++r) { mrow[r] = -1e30f; lrow[r] = 0.0f; }
#pragma unroll
  for (int t = 0; t < 4; ++t) acc[t] = v8f_zero();

  const int rbase = (lane & 16) ? 8 : 0;

  for (int s0 = 0; s0 < S_; s0 += 64) {
    __syncthreads();
    // ---- K tile via TDM (wave 0 issues; barrier publishes) ----
#if HAVE_TDM
    if (wid == 0) {
      tdm_load_2d_bf16(lds_offset_of(&Ks[0][0]),
                       Kw + kvbase + (size_t)s0 * HD, HD, 64,
                       (unsigned long long)HD, 4u, 3u);
    }
#endif
    // ---- V tile staged transposed [hd][key] by all threads ----
#pragma unroll
    for (int it = 0; it < 2; ++it) {
      int slot = tid + it * 256;        // 512 slots of 8 over 64x64
      int key = slot >> 3, c8 = slot & 7;
#if !HAVE_TDM
      uint4 dk = *(const uint4*)(Kw + kvbase + (size_t)(s0 + key) * HD + c8 * 8);
      *(uint4*)&Ks[key][c8 * 8] = dk;
#endif
      uint4 dv = *(const uint4*)(Vw + kvbase + (size_t)(s0 + key) * HD + c8 * 8);
      const unsigned short* e = (const unsigned short*)&dv;
#pragma unroll
      for (int j = 0; j < 8; ++j) Vt[c8 * 8 + j][key] = e[j];
    }
    if (tid < 64) inds[tid] = indic[(size_t)b * S_ + s0 + tid];
    if (s0 + 64 < S_) {
      __builtin_prefetch(Kw + kvbase + (size_t)(s0 + 64) * HD, 0, 0);
      __builtin_prefetch(Vw + kvbase + (size_t)(s0 + 64) * HD, 0, 0);
    }
#if HAVE_TDM
    if (wid == 0) __builtin_amdgcn_s_wait_tensorcnt(0);
#endif
    __syncthreads();

    // ---- scores: 4 key-tiles x (K=64 -> 2 WMMA) ----
    v8f sc[4];
#pragma unroll
    for (int kt = 0; kt < 4; ++kt) {
      sc[kt] = v8f_zero();
#pragma unroll
      for (int kc = 0; kc < 2; ++kc) {
        const unsigned short* bp = &Ks[kt * 16 + (lane & 15)][kc * 32 + bsel];
        v16bf bf = make_v16(*(const v8bf*)bp, *(const v8bf*)(bp + 16));
        sc[kt] = wmma_bf16(aq[kc], bf, sc[kt]);
      }
      sc[kt] = sc[kt] * 0.125f; // 1/sqrt(64)
    }

    // ---- online softmax (row stats live in registers, shfl over 16 lanes) --
    float mnew[8];
#pragma unroll
    for (int r = 0; r < 8; ++r) {
      float mx = fmaxf(fmaxf(sc[0][r], sc[1][r]), fmaxf(sc[2][r], sc[3][r]));
#pragma unroll
      for (int off = 1; off < 16; off <<= 1)
        mx = fmaxf(mx, __shfl_xor(mx, off, 32));
      mnew[r] = fmaxf(mrow[r], mx);
    }
    float rescale[8];
#pragma unroll
    for (int r = 0; r < 8; ++r) {
      rescale[r] = __expf(mrow[r] - mnew[r]);
      lrow[r] *= rescale[r];
      mrow[r] = mnew[r];
    }
#pragma unroll
    for (int t = 0; t < 4; ++t)
#pragma unroll
      for (int r = 0; r < 8; ++r) acc[t][r] *= rescale[r];

    float lacc[8];
#pragma unroll
    for (int r = 0; r < 8; ++r) lacc[r] = 0.0f;
#pragma unroll
    for (int kt = 0; kt < 4; ++kt) {
      float ind = inds[kt * 16 + (lane & 15)];
#pragma unroll
      for (int r = 0; r < 8; ++r) {
        float p = __expf(sc[kt][r] - mrow[r]);
        lacc[r] += p;                                    // UNMASKED denominator
        Ps[wid][rbase + r][kt * 16 + (lane & 15)] = f32_to_bf16_rne(p * ind);
      }
    }
#pragma unroll
    for (int r = 0; r < 8; ++r) {
      float ls = lacc[r];
#pragma unroll
      for (int off = 1; off < 16; off <<= 1)
        ls += __shfl_xor(ls, off, 32);
      lrow[r] += ls;
    }

    // ---- ctx += P * V : 2 key-chunks x 4 hd-tiles ----
#pragma unroll
    for (int c = 0; c < 2; ++c) {
      const unsigned short* pp = &Ps[wid][lane & 15][c * 32 + asel];
      v16bf pf = make_v16(*(const v8bf*)pp, *(const v8bf*)(pp + 16));
#pragma unroll
      for (int t = 0; t < 4; ++t) {
        const unsigned short* vp = &Vt[t * 16 + (lane & 15)][c * 32 + bsel];
        v16bf vf = make_v16(*(const v8bf*)vp, *(const v8bf*)(vp + 16));
        acc[t] = wmma_bf16(pf, vf, acc[t]);
      }
    }
  }

  // ---- normalize and write ctx as bf16 in [b, q, h*64+hd] layout ----
#pragma unroll
  for (int r = 0; r < 8; ++r) {
    float linv = 1.0f / lrow[r];
    int q = q0 + rbase + r;
    size_t o = ((size_t)b * NQ_ + q) * ((size_t)H_ * 64) + (size_t)h * 64;
#pragma unroll
    for (int t = 0; t < 4; ++t)
      ctxw[o + t * 16 + (lane & 15)] = f32_to_bf16_rne(acc[t][r] * linv);
  }
}

extern "C" void kernel_launch(void* const* d_in, const int* in_sizes, int n_in,
                              void* d_out, int out_size, void* d_ws, size_t ws_size,
                              hipStream_t stream) {
  (void)in_sizes; (void)n_in; (void)out_size; (void)ws_size;
  constexpr int B = 2, NQ = 2048, S = 2048, D = 1024, H = 16, HD = 64, OUT = 1024;

  const float* queries = (const float*)d_in[0];
  const float* keys    = (const float*)d_in[1];
  const float* values  = (const float*)d_in[2];
  const float* indic   = (const float*)d_in[3];
  const float* Wq      = (const float*)d_in[4];
  const float* Wk      = (const float*)d_in[5];
  const float* Wv      = (const float*)d_in[6];
  const float* Wo      = (const float*)d_in[7];

  // workspace layout (bf16): Q | K | V | ctx  (each 8.39 MB, total ~33.6 MB)
  unsigned short* Qw = (unsigned short*)d_ws;
  unsigned short* Kw = Qw + (size_t)B * H * NQ * HD;
  unsigned short* Vw = Kw + (size_t)B * H * S * HD;
  unsigned short* Cx = Vw + (size_t)B * H * S * HD;

  dim3 blk(256);

  // input projections: [B*len, D] x [D, H*HD] -> per-head bf16
  dim3 gproj((H * HD) / 128, (B * NQ) / 128);
  gemm_bf16_wmma<false, true><<<gproj, blk, 0, stream>>>(
      (const void*)queries, Wq, (void*)Qw, B * NQ, D, H * HD, NQ);
  gemm_bf16_wmma<false, true><<<gproj, blk, 0, stream>>>(
      (const void*)keys, Wk, (void*)Kw, B * S, D, H * HD, S);
  gemm_bf16_wmma<false, true><<<gproj, blk, 0, stream>>>(
      (const void*)values, Wv, (void*)Vw, B * S, D, H * HD, S);

  // attention
  dim3 gattn(B * H * (NQ / 128));
  flash_attn_bf16<<<gattn, blk, 0, stream>>>(Qw, Kw, Vw, indic, Cx, B, H, NQ, S);

  // output projection: [B*NQ, H*HD] x [H*HD, OUT] -> f32 out
  dim3 gout(OUT / 128, (B * NQ) / 128);
  gemm_bf16_wmma<true, false><<<gout, blk, 0, stream>>>(
      (const void*)Cx, Wo, d_out, B * NQ, H * HD, OUT, NQ);
}
